// _MegablocksAdapter_3040836846325
// MI455X (gfx1250) — matrix operand order repaired and checked by reading, not compiled, not run on hardware
//
#include <hip/hip_runtime.h>
#include <hip/hip_bf16.h>

// Problem dims: B=2, S=2048 -> T=4096 tokens, D=1024, FF=4096, E=8, top-2.
#define T_TOKENS 4096
#define DIM      1024
#define FFDIM    4096
#define NEXPERT  8

// GEMM tiling
#define BM 128
#define BN 128
#define BK 32
#define LDST 40   // LDS row stride in bf16 elements (80B rows: 64B data + 16B pad)

typedef __bf16 bf16x4  __attribute__((ext_vector_type(4)));
typedef __bf16 bf16x8  __attribute__((ext_vector_type(8)));
typedef __bf16 v16bf   __attribute__((ext_vector_type(16)));
typedef float  v8f     __attribute__((ext_vector_type(8)));
typedef int    i32x4   __attribute__((ext_vector_type(4)));
typedef int    i32x8   __attribute__((ext_vector_type(8)));
typedef unsigned int u32x4 __attribute__((ext_vector_type(4)));

#if defined(__HIP_DEVICE_COMPILE__) && __has_builtin(__builtin_amdgcn_tensor_load_to_lds)
#define HAVE_TDM 1
#endif
#if defined(__HIP_DEVICE_COMPILE__) && __has_builtin(__builtin_amdgcn_global_load_async_to_lds_b128)
#define HAVE_ASYNC_LDS 1
typedef __attribute__((address_space(1))) i32x4 gi32x4;   // global int4
typedef __attribute__((address_space(3))) i32x4 li32x4;   // LDS int4
#endif

#ifdef HAVE_TDM
// Issue one TDM 2D tile load: 128 rows x 64B from global (row stride K bf16
// elements) into LDS with 16B padding appended per row (-> 80B LDS stride).
// D# fields per CDNA5 ISA ch.8 (data_size=4B DWORDs).
__device__ __forceinline__ void tdm_load_tile(const __bf16* gsrc, const __bf16* ldst, int K) {
  const unsigned long long ga = (unsigned long long)(size_t)gsrc;
  const unsigned lds = (unsigned)(size_t)ldst;   // flat LDS addr: low 32 bits = LDS offset

  u32x4 g0;
  g0[0] = 1u;                                       // count=1 (valid), user descriptor
  g0[1] = lds;                                      // lds_addr
  g0[2] = (unsigned)(ga & 0xFFFFFFFFu);             // global_addr[31:0]
  g0[3] = (unsigned)((ga >> 32) & 0x01FFFFFFu)      // global_addr[56:32]
          | (2u << 30);                             // type = 2 ("image")

  const unsigned dim0 = (unsigned)(K >> 1);         // tensor_dim0 (DWORD units)
  const unsigned dim1 = 0x40000u;                   // tensor_dim1: ample rows (no OOB)
  i32x8 g1;
  g1[0] = (int)((2u << 16)                          // data_size = 4B
                | (1u << 20)                        // pad_enable
                | (3u << 22)                        // pad_interval: every 16 DWORDs (64B)
                | (3u << 25));                      // pad_amount: 4 DWORDs (16B)
  g1[1] = (int)((dim0 & 0xFFFFu) << 16);            // tensor_dim0[15:0] at bits 63:48
  g1[2] = (int)((dim0 >> 16) | ((dim1 & 0xFFFFu) << 16)); // dim0[31:16], dim1[15:0]
  g1[3] = (int)((dim1 >> 16) | (16u << 16));        // dim1[31:16], tile_dim0 = 16 DWORDs
  g1[4] = (int)128u;                                // tile_dim1 = 128 rows, tile_dim2 = 0
  g1[5] = (int)(unsigned)(K >> 1);                  // tensor_dim0_stride[31:0] (DWORDs)
  g1[6] = 0;                                        // stride0[47:32], stride1[15:0]
  g1[7] = 0;                                        // stride1[47:16]
  const i32x4 gz4 = {0, 0, 0, 0};                   // groups 2/3 unused (2D tile)
  const i32x8 gz8 = {0, 0, 0, 0, 0, 0, 0, 0};       // extra group (clang-23 6-arg form)
  __builtin_amdgcn_tensor_load_to_lds(g0, g1, gz4, gz4, gz8, 0);
}
#endif

// 16-byte global -> LDS copy (fallback staging path).
__device__ __forceinline__ void copy16_g2l(const __bf16* g, __bf16* l) {
#ifdef HAVE_ASYNC_LDS
  __builtin_amdgcn_global_load_async_to_lds_b128((gi32x4*)g, (li32x4*)l, 0, 0);
#else
  *(bf16x8*)l = *(const bf16x8*)g;
#endif
}

__device__ __forceinline__ float gelu_tanh(float x) {
  const float z = 0.7978845608028654f * (x + 0.044715f * x * x * x);
#if defined(__HIP_DEVICE_COMPILE__) && __has_builtin(__builtin_amdgcn_tanhf)
  return 0.5f * x * (1.0f + __builtin_amdgcn_tanhf(z));
#elif defined(__HIP_DEVICE_COMPILE__) && __has_builtin(__builtin_amdgcn_tanh_f32)
  return 0.5f * x * (1.0f + __builtin_amdgcn_tanh_f32(z));
#else
  // 0.5x(1+tanh(z)) == x*sigmoid(2z); branch-free, one hardware v_exp_f32.
  return x * (1.0f / (1.0f + __expf(-2.0f * z)));
#endif
}

// A-matrix 16x32 bf16 fragment (ISA 7.12.2): lanes 0-15 hold M=lane, K chunks {0..7,16..23};
// lanes 16-31 hold K chunks {8..15,24..31}. Two 16B LDS reads + register concat.
__device__ __forceinline__ v16bf load_frag_a(const __bf16* As, int row, int lane) {
  const __bf16* p = As + row * LDST;
  const int half = (lane >> 4) * 8;
  bf16x8 lo = *(const bf16x8*)(p + half);
  bf16x8 hi = *(const bf16x8*)(p + 16 + half);
  return __builtin_shufflevector(lo, hi, 0,1,2,3,4,5,6,7,8,9,10,11,12,13,14,15);
}

// B-matrix 32x16 bf16 fragment: lane holds column (lane&15), K chunks {0..7,16..23} (lanes 0-15) / {8..15,24..31} (lanes 16-31).
__device__ __forceinline__ v16bf load_frag_b(const __bf16* Bs, int nrow, int lane) {
  const __bf16* p = Bs + nrow * LDST + (lane >> 4) * 8;
  bf16x8 lo = *(const bf16x8*)(p);
  bf16x8 hi = *(const bf16x8*)(p + 16);
  return __builtin_shufflevector(lo, hi, 0,1,2,3,4,5,6,7,8,9,10,11,12,13,14,15);
}

// ---------------------------------------------------------------------------
// Router: probs = softmax(x @ router_w); top-2 -> dense gates[T][E]
// ---------------------------------------------------------------------------
__global__ __launch_bounds__(256)
void moe_router_kernel(const float* __restrict__ x,
                       const float* __restrict__ rw,
                       float* __restrict__ gates) {
  const int wave = threadIdx.x >> 5;
  const int lane = threadIdx.x & 31;
  const int tk = blockIdx.x * 8 + wave;
  if (tk >= T_TOKENS) return;

  float s[NEXPERT];
#pragma unroll
  for (int e = 0; e < NEXPERT; ++e) s[e] = 0.f;

  const float* xr = x + (size_t)tk * DIM;
  for (int d = lane; d < DIM; d += 32) {
    const float xv = xr[d];
    const float* w = rw + (size_t)d * NEXPERT;
#pragma unroll
    for (int e = 0; e < NEXPERT; ++e) s[e] += xv * w[e];
  }
#pragma unroll
  for (int off = 16; off > 0; off >>= 1) {
#pragma unroll
    for (int e = 0; e < NEXPERT; ++e) s[e] += __shfl_xor(s[e], off, 32);
  }
  float m = s[0];
#pragma unroll
  for (int e = 1; e < NEXPERT; ++e) m = fmaxf(m, s[e]);
  float p[NEXPERT], den = 0.f;
#pragma unroll
  for (int e = 0; e < NEXPERT; ++e) { p[e] = __expf(s[e] - m); den += p[e]; }
  const float inv = 1.f / den;
#pragma unroll
  for (int e = 0; e < NEXPERT; ++e) p[e] *= inv;
  int i1 = 0;
#pragma unroll
  for (int e = 1; e < NEXPERT; ++e) if (p[e] > p[i1]) i1 = e;
  int i2 = (i1 == 0) ? 1 : 0;
#pragma unroll
  for (int e = 0; e < NEXPERT; ++e) if (e != i1 && p[e] > p[i2]) i2 = e;

  if (lane < NEXPERT) {
    float g = (lane == i1) ? p[i1] : ((lane == i2) ? p[i2] : 0.f);
    gates[(size_t)tk * NEXPERT + lane] = g;
  }
}

// ---------------------------------------------------------------------------
// Elementwise fp32 -> bf16 (for x)
// ---------------------------------------------------------------------------
__global__ __launch_bounds__(256)
void convert_bf16_kernel(const float* __restrict__ src, __bf16* __restrict__ dst, int n) {
  const int i = (blockIdx.x * 256 + threadIdx.x) * 4;
  if (i + 3 < n) {
    const float4 v = *(const float4*)(src + i);
    bf16x4 b;
    b.x = (__bf16)v.x; b.y = (__bf16)v.y; b.z = (__bf16)v.z; b.w = (__bf16)v.w;
    *(bf16x4*)(dst + i) = b;
  }
}

// ---------------------------------------------------------------------------
// Tiled transpose + convert: src fp32 [R][C] (per expert) -> dst bf16 [C][R]
// ---------------------------------------------------------------------------
__global__ __launch_bounds__(256)
void transpose_convert_kernel(const float* __restrict__ src,
                              __bf16* __restrict__ dst,
                              int R, int C) {
  __shared__ float tile[64][65];
  const int t = threadIdx.x;
  const int r0 = blockIdx.y * 64;
  const int c0 = blockIdx.x * 64;
  const size_t eoff = (size_t)blockIdx.z * R * C;
#pragma unroll
  for (int i = 0; i < 16; ++i) {
    const int idx = i * 256 + t;         // 0..4095
    const int r = idx >> 6, c = idx & 63;
    tile[r][c] = src[eoff + (size_t)(r0 + r) * C + (c0 + c)];
  }
  __syncthreads();
#pragma unroll
  for (int i = 0; i < 16; ++i) {
    const int idx = i * 256 + t;
    const int cc = idx >> 6, rr = idx & 63;
    dst[eoff + (size_t)(c0 + cc) * R + (r0 + rr)] = (__bf16)tile[rr][cc];
  }
}

// ---------------------------------------------------------------------------
// Unified bf16 WMMA GEMM: C = A[M][K] x Bt[N][K]^T, double-buffered LDS,
// TDM (or async-LDS) staging overlapped with WMMA compute.
// MODE 0: H = bf16(gelu(C))         (GEMM1)
// MODE 1: out (+)= gates[:,e] * C   (GEMM2)
// ---------------------------------------------------------------------------
template <int MODE>
__global__ __launch_bounds__(256)
void moe_gemm_bf16_kernel(const __bf16* __restrict__ A,   // [M][K]
                          const __bf16* __restrict__ Bt,  // [N][K] (pre-transposed)
                          int K,
                          __bf16* __restrict__ H,         // MODE 0 output [M][FFDIM]
                          const float* __restrict__ gates,
                          float* __restrict__ out,        // MODE 1 output [M][DIM]
                          int expert, int first) {
  __shared__ __align__(16) __bf16 As[2][BM * LDST];
  __shared__ __align__(16) __bf16 Bs[2][BN * LDST];

  const int t    = threadIdx.x;
  const int lane = t & 31;
  const int wave = t >> 5;
  const int wm   = (wave & 1) * 64;   // 2 wave rows  (4 m-tiles each)
  const int wn   = (wave >> 1) * 32;  // 4 wave cols  (2 n-tiles each)
  const int bm   = blockIdx.y * BM;
  const int bn   = blockIdx.x * BN;
  const int nstages = K / BK;

  v8f acc[4][2];
  const v8f vzero = {0.f, 0.f, 0.f, 0.f, 0.f, 0.f, 0.f, 0.f};
#pragma unroll
  for (int mt = 0; mt < 4; ++mt)
#pragma unroll
    for (int nt = 0; nt < 2; ++nt) acc[mt][nt] = vzero;

  // Stage one BK-slice of both tiles into LDS buffer `buf`.
  auto stage = [&](int buf, int k0) {
#ifdef HAVE_TDM
    if (wave == 0) {
      tdm_load_tile(A  + (size_t)bm * K + k0, &As[buf][0], K);
      tdm_load_tile(Bt + (size_t)bn * K + k0, &Bs[buf][0], K);
    }
#else
#pragma unroll
    for (int j = 0; j < 2; ++j) {
      const int c   = t * 2 + j;          // 0..511
      const int row = c >> 2;
      const int seg = (c & 3) * 8;        // bf16 elements (16B granules)
      copy16_g2l(A  + (size_t)(bm + row) * K + k0 + seg, &As[buf][row * LDST + seg]);
      copy16_g2l(Bt + (size_t)(bn + row) * K + k0 + seg, &Bs[buf][row * LDST + seg]);
    }
#endif
  };

  auto wait_stage = [&]() {
#ifdef HAVE_TDM
    if (wave == 0) __builtin_amdgcn_s_wait_tensorcnt(0);
#elif defined(HAVE_ASYNC_LDS)
#if __has_builtin(__builtin_amdgcn_s_wait_asynccnt)
    __builtin_amdgcn_s_wait_asynccnt(0);
#else
    asm volatile("s_wait_asynccnt 0" ::: "memory");
#endif
#endif
  };

  stage(0, 0);
  wait_stage();
  __syncthreads();

  for (int s = 0; s < nstages; ++s) {
    const int cur = s & 1;
    if (s + 1 < nstages) stage(cur ^ 1, (s + 1) * BK);   // overlap with compute

    const __bf16* Ac = As[cur];
    const __bf16* Bc = Bs[cur];
    v16bf afr[4];
#pragma unroll
    for (int mt = 0; mt < 4; ++mt)
      afr[mt] = load_frag_a(Ac, wm + mt * 16 + (lane & 15), lane);
#pragma unroll
    for (int nt = 0; nt < 2; ++nt) {
      const v16bf bfr = load_frag_b(Bc, wn + nt * 16 + (lane & 15), lane);
#pragma unroll
      for (int mt = 0; mt < 4; ++mt) {
        acc[mt][nt] = __builtin_amdgcn_wmma_f32_16x16x32_bf16(
            false, afr[mt], false, bfr, (short)0, acc[mt][nt], false, false);
      }
    }
    wait_stage();
    __syncthreads();
  }

  // C/D layout: elem i, lane l -> M = i + 8*(l>>4), N = l&15.
  if (MODE == 0) {
#pragma unroll
    for (int mt = 0; mt < 4; ++mt)
#pragma unroll
      for (int nt = 0; nt < 2; ++nt)
#pragma unroll
        for (int i = 0; i < 8; ++i) {
          const int row = bm + wm + mt * 16 + i + 8 * (lane >> 4);
          const int col = bn + wn + nt * 16 + (lane & 15);
          H[(size_t)row * FFDIM + col] = (__bf16)gelu_tanh(acc[mt][nt][i]);
        }
  } else {
#pragma unroll
    for (int mt = 0; mt < 4; ++mt)
#pragma unroll
      for (int i = 0; i < 8; ++i) {
        const int row = bm + wm + mt * 16 + i + 8 * (lane >> 4);
        const float g = gates[(size_t)row * NEXPERT + expert];
#pragma unroll
        for (int nt = 0; nt < 2; ++nt) {
          const int col = bn + wn + nt * 16 + (lane & 15);
          const size_t oi = (size_t)row * DIM + col;
          const float prev = first ? 0.f : out[oi];
          out[oi] = prev + g * acc[mt][nt][i];
        }
      }
  }
}

// ---------------------------------------------------------------------------
// Host launcher
// ---------------------------------------------------------------------------
extern "C" void kernel_launch(void* const* d_in, const int* in_sizes, int n_in,
                              void* d_out, int out_size, void* d_ws, size_t ws_size,
                              hipStream_t stream) {
  const float* x        = (const float*)d_in[0];   // [T,D]
  const float* router_w = (const float*)d_in[1];   // [D,E]
  const float* w1       = (const float*)d_in[2];   // [E,D,FF]
  const float* w2       = (const float*)d_in[3];   // [E,FF,D]
  float* out = (float*)d_out;                      // [T,D]

  // Workspace layout: gates fp32 | xb bf16 | w1t bf16 | w2t bf16 | h bf16
  float*  gates = (float*)d_ws;
  __bf16* xb    = (__bf16*)((char*)d_ws + (size_t)T_TOKENS * NEXPERT * sizeof(float));
  __bf16* w1t   = xb  + (size_t)T_TOKENS * DIM;                 // [E][FF][D]
  __bf16* w2t   = w1t + (size_t)NEXPERT * DIM * FFDIM;          // [E][D][FF]
  __bf16* h     = w2t + (size_t)NEXPERT * DIM * FFDIM;          // [T][FF]

  const dim3 blk(256);

  moe_router_kernel<<<T_TOKENS / 8, blk, 0, stream>>>(x, router_w, gates);
  convert_bf16_kernel<<<(T_TOKENS * DIM / 4 + 255) / 256, blk, 0, stream>>>(
      x, xb, T_TOKENS * DIM);
  // w1 [E][D][FF] -> w1t [E][FF][D];  w2 [E][FF][D] -> w2t [E][D][FF]
  transpose_convert_kernel<<<dim3(FFDIM / 64, DIM / 64, NEXPERT), blk, 0, stream>>>(
      w1, w1t, DIM, FFDIM);
  transpose_convert_kernel<<<dim3(DIM / 64, FFDIM / 64, NEXPERT), blk, 0, stream>>>(
      w2, w2t, FFDIM, DIM);

  const dim3 g1(FFDIM / BN, T_TOKENS / BM);  // 32 x 32
  const dim3 g2(DIM / BN, T_TOKENS / BM);    //  8 x 32
  for (int e = 0; e < NEXPERT; ++e) {
    moe_gemm_bf16_kernel<0><<<g1, blk, 0, stream>>>(
        xb, w1t + (size_t)e * DIM * FFDIM, DIM, h, nullptr, nullptr, 0, 0);
    moe_gemm_bf16_kernel<1><<<g2, blk, 0, stream>>>(
        h, w2t + (size_t)e * DIM * FFDIM, FFDIM, nullptr, gates, out, e, (e == 0) ? 1 : 0);
  }
}
